// AdditiveAttention_26216480375402
// MI455X (gfx1250) — matrix operand order repaired and checked by reading, not compiled, not run on hardware
//
#include <hip/hip_runtime.h>
#include <hip/hip_bf16.h>
#include <math.h>

typedef __attribute__((ext_vector_type(16))) __bf16 bf16x16;
typedef __attribute__((ext_vector_type(8)))  __bf16 bf16x8;
typedef __attribute__((ext_vector_type(8)))  float  f32x8;

#define L_DIM 1024
#define D_DIM 256
#define B_DIM 16
#define N_DIM 4096
#define NEG_BIG (-1e9f)

// ---------------- setup kernels ----------------

// W1 [L=1024][D=256] f32 -> W1t [D][L] bf16 (coalesced read, scattered write)
__global__ __launch_bounds__(256) void k_convert_w1t(const float* __restrict__ W1,
                                                     __bf16* __restrict__ W1t) {
  int idx = blockIdx.x * 256 + threadIdx.x;   // 0..262143
  int k = idx >> 8;                           // 0..1023 (L index)
  int n = idx & 255;                          // 0..255  (D index)
  W1t[(size_t)n * L_DIM + k] = (__bf16)W1[idx];
}

// W2 [D][D] f32 -> W2t [outD][inD] bf16
__global__ __launch_bounds__(256) void k_convert_w2t(const float* __restrict__ W2,
                                                     __bf16* __restrict__ W2t) {
  int idx = blockIdx.x * 256 + threadIdx.x;   // 0..65535
  int k = idx >> 8;
  int n = idx & 255;
  W2t[(size_t)n * D_DIM + k] = (__bf16)W2[idx];
}

// fold BN into per-channel scale/shift: y = h*s + t,  h = x@W1 (b1 folded into t)
__global__ __launch_bounds__(256) void k_bn(const float* b1, const float* gam,
                                            const float* bet, const float* mean,
                                            const float* var, float* sArr, float* tArr) {
  int d = threadIdx.x;
  float s = gam[d] * rsqrtf(var[d] + 1e-5f);
  sArr[d] = s;
  tArr[d] = (b1[d] - mean[d]) * s + bet[d];
}

// scores := b3 (atomic accumulation target), M := 0
__global__ __launch_bounds__(256) void k_init(const float* b3, float* scores, float* Mbuf) {
  int idx = blockIdx.x * 256 + threadIdx.x;   // 320 blocks -> 81920 == 65536 + 16384
  if (idx < B_DIM * N_DIM) scores[idx] = b3[0];
  else Mbuf[idx - B_DIM * N_DIM] = 0.f;
}

// ---------------- fused attention-MLP mega kernel ----------------
// Per block: 128 rows of x (flattened [B*N, L]).
// GEMM1 (x@W1 + BN + leaky) -> LDS bf16 tile; GEMM2 (@W2 + leaky) -> scores (@W3, atomic).
__global__ __launch_bounds__(256) void k_mega(const float* __restrict__ x,
                                              const __bf16* __restrict__ W1t,
                                              const __bf16* __restrict__ W2t,
                                              const float* __restrict__ sArr,
                                              const float* __restrict__ tArr,
                                              const float* __restrict__ b2,
                                              const float* __restrict__ W3,
                                              float* __restrict__ scores) {
  extern __shared__ char smem[];
  __bf16* xs = (__bf16*)smem;                       // [128][40]  (k-step staging, padded)
  __bf16* h1 = (__bf16*)(smem + 128 * 40 * 2);      // [128][264] (hidden tile, padded)
  const int XS_STR = 40, H1_STR = 264;

  const int tid  = threadIdx.x;
  const int lane = tid & 31;
  const int wave = tid >> 5;
  const int mw   = (wave >> 2) * 64;   // wave row offset   (2 wave-rows)
  const int nw   = (wave & 3) * 64;    // wave col offset   (4 wave-cols)
  const int lcol = lane & 15;
  const int half = lane >> 4;          // 0: lanes 0-15, 1: lanes 16-31
  const size_t mbase = (size_t)blockIdx.x * 128;

  f32x8 acc[4][4];
  #pragma unroll
  for (int i = 0; i < 4; ++i)
    #pragma unroll
    for (int j = 0; j < 4; ++j)
      #pragma unroll
      for (int r = 0; r < 8; ++r) acc[i][j][r] = 0.f;

  // ---------- GEMM1: K = 1024 ----------
  for (int ks = 0; ks < L_DIM; ks += 32) {
    __syncthreads();
    {   // stage 128x32 fp32 -> bf16 (2 threads per row, 16 elems each)
      int row = tid >> 1, h2 = tid & 1;
      const float4* gp = (const float4*)(x + (mbase + row) * L_DIM + ks + h2 * 16);
      float4 f0 = gp[0], f1 = gp[1], f2 = gp[2], f3 = gp[3];
      bf16x8 p0 = {(__bf16)f0.x, (__bf16)f0.y, (__bf16)f0.z, (__bf16)f0.w,
                   (__bf16)f1.x, (__bf16)f1.y, (__bf16)f1.z, (__bf16)f1.w};
      bf16x8 p1 = {(__bf16)f2.x, (__bf16)f2.y, (__bf16)f2.z, (__bf16)f2.w,
                   (__bf16)f3.x, (__bf16)f3.y, (__bf16)f3.z, (__bf16)f3.w};
      bf16x8* lp = (bf16x8*)(xs + row * XS_STR + h2 * 16);
      lp[0] = p0; lp[1] = p1;
    }
    __syncthreads();

    // B fragments from global bf16 W1t (L2-resident). Lane n = column n,
    // 16 consecutive K starting at half*16.
    bf16x16 bfrag[4];
    #pragma unroll
    for (int tn = 0; tn < 4; ++tn) {
      const bf16x8* gp = (const bf16x8*)(W1t + (size_t)(nw + tn * 16 + lcol) * L_DIM + ks);
      union { bf16x16 v; bf16x8 h[2]; } u;
      u.h[0] = gp[half * 2];
      u.h[1] = gp[half * 2 + 1];
      bfrag[tn] = u.v;
    }
    // A fragments from LDS: lanes 0-15 K{0..15}, lanes 16-31 K{16..31}
    #pragma unroll
    for (int tm = 0; tm < 4; ++tm) {
      const bf16x8* lp = (const bf16x8*)(xs + (mw + tm * 16 + lcol) * XS_STR);
      union { bf16x16 v; bf16x8 h[2]; } u;
      u.h[0] = lp[half * 2];
      u.h[1] = lp[half * 2 + 1];
      #pragma unroll
      for (int tn = 0; tn < 4; ++tn)
        acc[tm][tn] = __builtin_amdgcn_wmma_f32_16x16x32_bf16(
            false, u.v, false, bfrag[tn], (short)0, acc[tm][tn], false, false);
    }
  }

  // ---------- epilogue 1: BN + leaky -> LDS h1 (bf16) ----------
  #pragma unroll
  for (int tn = 0; tn < 4; ++tn) {
    int col = nw + tn * 16 + lcol;
    float s = sArr[col], t = tArr[col];
    #pragma unroll
    for (int tm = 0; tm < 4; ++tm) {
      int rbase = mw + tm * 16 + half * 8;
      #pragma unroll
      for (int r = 0; r < 8; ++r) {
        float v = acc[tm][tn][r] * s + t;
        v = (v >= 0.f) ? v : 0.2f * v;
        h1[(rbase + r) * H1_STR + col] = (__bf16)v;
      }
    }
  }
  __syncthreads();

  // ---------- GEMM2: K = 256 ----------
  #pragma unroll
  for (int i = 0; i < 4; ++i)
    #pragma unroll
    for (int j = 0; j < 4; ++j)
      #pragma unroll
      for (int r = 0; r < 8; ++r) acc[i][j][r] = 0.f;

  for (int ks = 0; ks < D_DIM; ks += 32) {
    bf16x16 bfrag[4];
    #pragma unroll
    for (int tn = 0; tn < 4; ++tn) {
      const bf16x8* gp = (const bf16x8*)(W2t + (size_t)(nw + tn * 16 + lcol) * D_DIM + ks);
      union { bf16x16 v; bf16x8 h[2]; } u;
      u.h[0] = gp[half * 2];
      u.h[1] = gp[half * 2 + 1];
      bfrag[tn] = u.v;
    }
    #pragma unroll
    for (int tm = 0; tm < 4; ++tm) {
      const bf16x8* lp = (const bf16x8*)(h1 + (size_t)(mw + tm * 16 + lcol) * H1_STR + ks);
      union { bf16x16 v; bf16x8 h[2]; } u;
      u.h[0] = lp[half * 2];
      u.h[1] = lp[half * 2 + 1];
      #pragma unroll
      for (int tn = 0; tn < 4; ++tn)
        acc[tm][tn] = __builtin_amdgcn_wmma_f32_16x16x32_bf16(
            false, u.v, false, bfrag[tn], (short)0, acc[tm][tn], false, false);
    }
  }

  // ---------- epilogue 2: +b2, leaky, dot W3, half-wave reduce, atomic scores ----------
  float w3v[4], b2v[4];
  #pragma unroll
  for (int tn = 0; tn < 4; ++tn) {
    int col = nw + tn * 16 + lcol;
    w3v[tn] = W3[col];
    b2v[tn] = b2[col];
  }
  #pragma unroll
  for (int tm = 0; tm < 4; ++tm) {
    #pragma unroll
    for (int r = 0; r < 8; ++r) {
      float p = 0.f;
      #pragma unroll
      for (int tn = 0; tn < 4; ++tn) {
        float v = acc[tm][tn][r] + b2v[tn];
        v = (v >= 0.f) ? v : 0.2f * v;
        p += v * w3v[tn];
      }
      #pragma unroll
      for (int m = 1; m < 16; m <<= 1) p += __shfl_xor(p, m, 32);
      if (lcol == 0)
        atomicAdd(&scores[mbase + mw + tm * 16 + r + half * 8], p);
    }
  }
}

// ---------------- masked softmax over N per batch ----------------
__global__ __launch_bounds__(256) void k_softmax(const float* __restrict__ scores,
                                                 const int* __restrict__ mask,
                                                 float* __restrict__ A) {
  __shared__ float red[256];
  int b = blockIdx.x, tid = threadIdx.x;
  const float* sp = scores + (size_t)b * N_DIM;
  const int*   mp = mask + (size_t)b * N_DIM;
  float mx = NEG_BIG;
  for (int n = tid; n < N_DIM; n += 256) {
    float s = mp[n] ? sp[n] : NEG_BIG;
    mx = fmaxf(mx, s);
  }
  red[tid] = mx; __syncthreads();
  for (int st = 128; st > 0; st >>= 1) {
    if (tid < st) red[tid] = fmaxf(red[tid], red[tid + st]);
    __syncthreads();
  }
  float gm = red[0]; __syncthreads();
  float* Ap = A + (size_t)b * N_DIM;
  float sum = 0.f;
  for (int n = tid; n < N_DIM; n += 256) {
    float e = mp[n] ? expf(sp[n] - gm) : 0.f;
    Ap[n] = e; sum += e;
  }
  red[tid] = sum; __syncthreads();
  for (int st = 128; st > 0; st >>= 1) {
    if (tid < st) red[tid] += red[tid + st];
    __syncthreads();
  }
  float inv = 1.f / red[0];
  for (int n = tid; n < N_DIM; n += 256) Ap[n] *= inv;
}

// ---------------- attention pool M[b,l] = sum_n A[b,n]*x[b,n,l] ----------------
__global__ __launch_bounds__(256) void k_pool(const float* __restrict__ x,
                                              const float* __restrict__ A,
                                              float* __restrict__ M) {
  int l = blockIdx.x * 256 + threadIdx.x;
  int b = blockIdx.z;
  int n0 = blockIdx.y * 256;
  const float* xp = x + ((size_t)(b * N_DIM + n0)) * L_DIM + l;
  const float* Ap = A + (size_t)b * N_DIM + n0;
  float acc = 0.f;
  for (int i = 0; i < 256; ++i) {
    float a = Ap[i];                  // uniform per block -> scalar branch
    if (a != 0.f) acc += a * xp[(size_t)i * L_DIM];
  }
  atomicAdd(&M[(size_t)b * L_DIM + l], acc);
}

// ---------------- tiny classifier: M@C1 relu @C2 relu @C3 softmax/argmax ----------------
__global__ __launch_bounds__(256) void k_cls(const float* __restrict__ M,
                                             const float* C1, const float* c1,
                                             const float* C2, const float* c2,
                                             const float* C3, const float* c3,
                                             float* __restrict__ out) {
  extern __shared__ char smem[];
  float* Ml = (float*)smem;            // 16*1024
  float* z1 = Ml + 16 * 1024;          // 16*256
  float* z2 = z1 + 16 * 256;           // 16*256
  float* lg = z2 + 16 * 256;           // 32
  int tid = threadIdx.x;
  for (int i = tid; i < 16 * 1024; i += 256) Ml[i] = M[i];
  __syncthreads();
  int r = tid >> 4, dg = tid & 15;
  for (int j = 0; j < 16; ++j) {
    int d = dg * 16 + j;
    float a = c1[d];
    const float* mr = Ml + r * 1024;
    for (int l = 0; l < 1024; ++l) a += mr[l] * C1[l * 256 + d];
    z1[r * 256 + d] = fmaxf(a, 0.f);
  }
  __syncthreads();
  for (int j = 0; j < 16; ++j) {
    int d = dg * 16 + j;
    float a = c2[d];
    const float* zr = z1 + r * 256;
    for (int e = 0; e < 256; ++e) a += zr[e] * C2[e * 256 + d];
    z2[r * 256 + d] = fmaxf(a, 0.f);
  }
  __syncthreads();
  if (tid < 32) {
    int rr = tid >> 1, jj = tid & 1;
    float a = c3[jj];
    const float* zr = z2 + rr * 256;
    for (int d = 0; d < 256; ++d) a += zr[d] * C3[d * 2 + jj];
    lg[tid] = a;
  }
  __syncthreads();
  if (tid < 16) {
    float l0 = lg[tid * 2], l1 = lg[tid * 2 + 1];
    float m = fmaxf(l0, l1);
    float e0 = expf(l0 - m), e1 = expf(l1 - m);
    float s = e0 + e1;
    out[tid * 2]     = e0 / s;          // Y_prob
    out[tid * 2 + 1] = e1 / s;
    out[32 + tid] = (l1 > l0) ? 1.0f : 0.0f;   // Y_hat (argmax, first-max tie rule)
  }
}

extern "C" void kernel_launch(void* const* d_in, const int* in_sizes, int n_in,
                              void* d_out, int out_size, void* d_ws, size_t ws_size,
                              hipStream_t stream) {
  const float* x    = (const float*)d_in[0];
  const int*   mask = (const int*)d_in[1];
  const float* W1   = (const float*)d_in[2];
  const float* b1   = (const float*)d_in[3];
  const float* gam  = (const float*)d_in[4];
  const float* bet  = (const float*)d_in[5];
  const float* mean = (const float*)d_in[6];
  const float* var  = (const float*)d_in[7];
  const float* W2   = (const float*)d_in[8];
  const float* b2   = (const float*)d_in[9];
  const float* W3   = (const float*)d_in[10];
  const float* b3   = (const float*)d_in[11];
  const float* C1   = (const float*)d_in[12];
  const float* c1   = (const float*)d_in[13];
  const float* C2   = (const float*)d_in[14];
  const float* c2   = (const float*)d_in[15];
  const float* C3   = (const float*)d_in[16];
  const float* c3   = (const float*)d_in[17];

  char* ws = (char*)d_ws;
  __bf16* W1t   = (__bf16*)ws; ws += (size_t)L_DIM * D_DIM * 2;   // 512 KB
  __bf16* W2t   = (__bf16*)ws; ws += (size_t)D_DIM * D_DIM * 2;   // 128 KB
  float* sArr   = (float*)ws;  ws += D_DIM * 4;
  float* tArr   = (float*)ws;  ws += D_DIM * 4;
  float* scores = (float*)ws;  ws += (size_t)B_DIM * N_DIM * 4;   // 256 KB
  float* Aat    = (float*)ws;  ws += (size_t)B_DIM * N_DIM * 4;   // 256 KB
  float* Mbuf   = (float*)ws;  ws += (size_t)B_DIM * L_DIM * 4;   //  64 KB

  k_convert_w1t<<<1024, 256, 0, stream>>>(W1, W1t);
  k_convert_w2t<<<256, 256, 0, stream>>>(W2, W2t);
  k_bn<<<1, 256, 0, stream>>>(b1, gam, bet, mean, var, sArr, tArr);
  k_init<<<320, 256, 0, stream>>>(b3, scores, Mbuf);

  size_t megaLds = (size_t)(128 * 40 + 128 * 264) * sizeof(__bf16);   // ~76 KB
  k_mega<<<512, 256, megaLds, stream>>>(x, W1t, W2t, sArr, tArr, b2, W3, scores);

  k_softmax<<<16, 256, 0, stream>>>(scores, mask, Aat);
  k_pool<<<dim3(4, 16, 16), 256, 0, stream>>>(x, Aat, Mbuf);

  size_t clsLds = (size_t)(16 * 1024 + 16 * 256 * 2 + 32) * sizeof(float); // ~96 KB
  k_cls<<<1, 256, clsLds, stream>>>(Mbuf, C1, c1, C2, c2, C3, c3, (float*)d_out);
}
